// GQA_Flash_RoPE_13958643712131
// MI455X (gfx1250) — matrix operand order repaired and checked by reading, not compiled, not run on hardware
//
#include <hip/hip_runtime.h>
#include <hip/hip_bf16.h>

typedef __attribute__((ext_vector_type(16))) __bf16 v16bf;
typedef __attribute__((ext_vector_type(8)))  __bf16 v8bf;
typedef __attribute__((ext_vector_type(8)))  float  v8f;

#define B_   2
#define T_   2048
#define D_   2048
#define HQ_  32
#define HKV_ 8
#define DK_  64
#define G_   4
#define BT_  (B_*T_)
#define DKV_ (HKV_*DK_)   // 512

// ---------- WMMA helpers ----------
__device__ __forceinline__ v8f wmma_bf16(v16bf a, v16bf b, v8f c) {
  return __builtin_amdgcn_wmma_f32_16x16x32_bf16(false, a, false, b, (short)0, c, false, false);
}

// A fragment (16x32 bf16): per lane (row m = lane&15):
//   elems 0..7  = A[m][k0 + half*16 .. +7]       (VGPRs 0..3)
//   elems 8..15 = A[m][k0 + half*16 + 8 .. +7]   (VGPRs 4..7)
__device__ __forceinline__ v16bf load_a16(const __bf16* row, int half) {
  union { v16bf v; v8bf h[2]; } u;
  u.h[0] = *(const v8bf*)(row + half * 16);
  u.h[1] = *(const v8bf*)(row + half * 16 + 8);
  return u.v;
}

// ---------- conversion kernels ----------
__global__ void f32_to_bf16(const float* __restrict__ in, __bf16* __restrict__ out, int n) {
  int i = blockIdx.x * blockDim.x + threadIdx.x;
  if (i < n) out[i] = (__bf16)in[i];
}

// in: K x N (f32 row-major)  ->  out: N x K (bf16, i.e. B transposed)
__global__ void transpose_to_bf16(const float* __restrict__ in, __bf16* __restrict__ out,
                                  int K, int N) {
  int i = blockIdx.x * blockDim.x + threadIdx.x;
  if (i < K * N) {
    int n = i % N, k = i / N;
    out[(size_t)n * K + k] = (__bf16)in[i];
  }
}

// ---------- WMMA GEMM with async-LDS B staging ----------
// C(MxN,f32) = A(MxK,bf16) @ B, with B given as Bt (NxK,bf16).
// block = 256 thr = 8 waves; wave -> 16x64 tile; block -> 128x64 tile.
// B tile (64n x 32k) is staged once per block into LDS via
// global_load_async_to_lds_b128 (ASYNCcnt), double-buffered so the DMA for
// tile k+1 overlaps the WMMAs of tile k. A fragments are register
// double-buffered.
__global__ __launch_bounds__(256) void gemm_bf16_f32(
    const __bf16* __restrict__ A, const __bf16* __restrict__ Bt,
    float* __restrict__ C, int M, int N, int K)
{
  __shared__ __bf16 btile[2][64][32];   // 8 KB double-buffered B tile
  const int tid  = threadIdx.x;
  const int wave = tid >> 5, lane = tid & 31;
  const int half = lane >> 4, lm = lane & 15;
  const int row0 = blockIdx.x * 128 + wave * 16;
  const int col0 = blockIdx.y * 64;
  const __bf16* arow = A + (size_t)(row0 + lm) * K;

  // async staging assignment: thread t moves 16B; n = t>>2, 16B-chunk = t&3
  const int ln = tid >> 2, lc = tid & 3;
  const __bf16* gsrc = Bt + (size_t)(col0 + ln) * K + lc * 8;
  unsigned lds0 = (unsigned)(size_t)&btile[0][ln][lc * 8];
  unsigned lds1 = (unsigned)(size_t)&btile[1][ln][lc * 8];

#define ASYNC_B(ldsoff, k0)                                                   \
  do {                                                                        \
    unsigned long long ga_ = (unsigned long long)(const void*)(gsrc + (k0));  \
    asm volatile("global_load_async_to_lds_b128 %0, %1, off"                  \
                 :: "v"(ldsoff), "v"(ga_) : "memory");                        \
  } while (0)

  v8f acc[4] = {};
  ASYNC_B(lds0, 0);
  asm volatile("s_wait_asynccnt 0" ::: "memory");
  __syncthreads();

  v16bf a_cur = load_a16(arow, half);
  int buf = 0;
  for (int k0 = 0; k0 < K; k0 += 32) {
    const bool more = (k0 + 32) < K;
    // kick off next B tile DMA + next A fragment loads before computing
    if (more) {
      if (buf == 0) ASYNC_B(lds1, k0 + 32); else ASYNC_B(lds0, k0 + 32);
      __builtin_prefetch(arow + k0 + 256, 0, 1);
    }
    v16bf a_nxt = a_cur;
    if (more) a_nxt = load_a16(arow + k0 + 32, half);
#pragma unroll
    for (int nt = 0; nt < 4; ++nt) {
      // B fragment: lane column n = nt*16+lm; elems = btile[buf][n][half*16 .. +15]
      v16bf b = *(const v16bf*)&btile[buf][nt * 16 + lm][half * 16];
      acc[nt] = wmma_bf16(a_cur, b, acc[nt]);
    }
    asm volatile("s_wait_asynccnt 0" ::: "memory");
    __syncthreads();
    a_cur = a_nxt;
    buf ^= 1;
  }
#undef ASYNC_B

#pragma unroll
  for (int nt = 0; nt < 4; ++nt)
#pragma unroll
    for (int r = 0; r < 8; ++r)
      C[(size_t)(row0 + r + 8 * half) * N + col0 + nt * 16 + lm] = acc[nt][r];
}

// ---------- RoPE: in [B,T,H,DK] f32 -> out [B,H,T,DK] bf16 ----------
__global__ void rope_convert(const float* __restrict__ in, __bf16* __restrict__ out, int H) {
  int tid = blockIdx.x * blockDim.x + threadIdx.x;
  int total = B_ * H * T_ * (DK_ / 2);
  if (tid >= total) return;
  int i = tid & 31;                 // pair index 0..31
  int t = (tid >> 5) % T_;
  int h = ((tid >> 5) / T_) % H;
  int b = tid / (32 * T_ * H);
  const float* src = in + (((size_t)b * T_ + t) * H + h) * DK_ + 2 * i;
  float e = src[0], o = src[1];
  float freq = __powf(10000.f, -(float)(2 * i) / (float)DK_);
  float ang = (float)t * freq;
  float sn, cs;
  __sincosf(ang, &sn, &cs);
  __bf16* dst = out + (((size_t)b * H + h) * T_ + t) * DK_ + 2 * i;
  dst[0] = (__bf16)(e * cs - o * sn);
  dst[1] = (__bf16)(o * cs + e * sn);
}

// ---------- V transpose: in [B,T,HKV,DK] f32 -> out [B,HKV,DK,T] bf16 ----------
__global__ void v_transpose(const float* __restrict__ in, __bf16* __restrict__ out) {
  int tid = blockIdx.x * blockDim.x + threadIdx.x;
  int total = B_ * HKV_ * DK_ * T_;
  if (tid >= total) return;
  int t = tid % T_;
  int d = (tid / T_) % DK_;
  int h = (tid / (T_ * DK_)) % HKV_;
  int b = tid / (T_ * DK_ * HKV_);
  out[tid] = (__bf16)in[(((size_t)b * T_ + t) * HKV_ + h) * DK_ + d];
}

// ---------- Flash attention: one wave per (b, hq, 16 q rows); KV tiles of 32 ----------
// Q  [B,HQ,T,DK] bf16 (post-RoPE), Kh [B,HKV,T,DK] bf16 (post-RoPE),
// Vt [B,HKV,DK,T] bf16, O [B,T,HQ,DK] bf16.
__global__ __launch_bounds__(256) void attn_fwd(
    const __bf16* __restrict__ Q, const __bf16* __restrict__ Kh,
    const __bf16* __restrict__ Vt, __bf16* __restrict__ O)
{
  __shared__ __bf16 lds_p[8][16][32];         // per-wave P tile (16 q x 32 kv)
  const int wave = threadIdx.x >> 5, lane = threadIdx.x & 31;
  const int half = lane >> 4, lm = lane & 15;
  const int tilesPerHead = T_ / 16;
  int gw = blockIdx.x * 8 + wave;
  int qt = gw % tilesPerHead;
  int head = gw / tilesPerHead;               // 0 .. B*HQ-1
  int hq = head % HQ_, b = head / HQ_;
  int hkv = hq / G_;
  int q0 = qt * 16;

  const __bf16* qrow  = Q  + (((size_t)b * HQ_ + hq) * T_ + q0 + lm) * DK_;
  const __bf16* kbase = Kh + ((size_t)b * HKV_ + hkv) * T_ * DK_;
  const __bf16* vbase = Vt + ((size_t)b * HKV_ + hkv) * (size_t)DK_ * T_;

  v16bf qa0 = load_a16(qrow, half);           // K dim 0..31
  v16bf qa1 = load_a16(qrow + 32, half);      // K dim 32..63

  float mrow[8], lrow[8];
  v8f o0 = {}, o1 = {}, o2 = {}, o3 = {};
#pragma unroll
  for (int r = 0; r < 8; ++r) { mrow[r] = -1e30f; lrow[r] = 0.f; }
  const float scale = 0.125f;                 // 1/sqrt(64)

  for (int kv0 = 0; kv0 <= q0 + 15; kv0 += 32) {
    // ---- scores S = Q @ K^T for kv columns kv0..kv0+31 (two 16-col sub-tiles) ----
    v8f s0 = {}, s1 = {};
    {
      const __bf16* kr = kbase + (size_t)(kv0 + lm) * DK_;
      s0 = wmma_bf16(qa0, *(const v16bf*)(kr + half * 16), s0);
      s0 = wmma_bf16(qa1, *(const v16bf*)(kr + 32 + half * 16), s0);
    }
    {
      const __bf16* kr = kbase + (size_t)(kv0 + 16 + lm) * DK_;
      s1 = wmma_bf16(qa0, *(const v16bf*)(kr + half * 16), s1);
      s1 = wmma_bf16(qa1, *(const v16bf*)(kr + 32 + half * 16), s1);
    }
    // ---- online softmax (rows m = r + 8*half, kv col = lm / 16+lm) ----
#pragma unroll
    for (int r = 0; r < 8; ++r) {
      int qi = q0 + r + 8 * half;
      float va = ((kv0 + lm) <= qi)      ? s0[r] * scale : -1e30f;
      float vb = ((kv0 + 16 + lm) <= qi) ? s1[r] * scale : -1e30f;
      float mt = fmaxf(va, vb);
#pragma unroll
      for (int off = 1; off < 16; off <<= 1) mt = fmaxf(mt, __shfl_xor(mt, off, 16));
      float mn = fmaxf(mrow[r], mt);
      float alpha = __expf(mrow[r] - mn);
      mrow[r] = mn;
      float pa = __expf(va - mn), pb = __expf(vb - mn);
      float rs = pa + pb;
#pragma unroll
      for (int off = 1; off < 16; off <<= 1) rs += __shfl_xor(rs, off, 16);
      lrow[r] = lrow[r] * alpha + rs;
      o0[r] *= alpha; o1[r] *= alpha; o2[r] *= alpha; o3[r] *= alpha;
      lds_p[wave][r + 8 * half][lm]      = (__bf16)pa;
      lds_p[wave][r + 8 * half][16 + lm] = (__bf16)pb;
    }
    // D-layout -> A-layout via per-wave LDS (in-order DS pipe; block compiler reordering)
    asm volatile("s_wait_dscnt 0" ::: "memory");
    __builtin_amdgcn_wave_barrier();
    v16bf pfrag;
    {
      const __bf16* prow = &lds_p[wave][lm][0];
#pragma unroll
      for (int j = 0; j < 8; ++j) {
        int k = half * 16 + j * 2;
        pfrag[2 * j]     = prow[k];
        pfrag[2 * j + 1] = prow[k + 1];
      }
    }
    __builtin_amdgcn_wave_barrier();
    // ---- O += P @ V ; V B-fragment: col n = nt*16+lm, elems Vt[n][kv0 + half*16 ..+15] ----
    const __bf16* vr = vbase + (size_t)lm * T_ + kv0 + half * 16;
    o0 = wmma_bf16(pfrag, *(const v16bf*)(vr), o0);
    o1 = wmma_bf16(pfrag, *(const v16bf*)(vr + 16 * (size_t)T_), o1);
    o2 = wmma_bf16(pfrag, *(const v16bf*)(vr + 32 * (size_t)T_), o2);
    o3 = wmma_bf16(pfrag, *(const v16bf*)(vr + 48 * (size_t)T_), o3);
  }

#pragma unroll
  for (int r = 0; r < 8; ++r) {
    float inv = 1.f / lrow[r];
    int qi = q0 + r + 8 * half;
    __bf16* orow = O + (((size_t)b * T_ + qi) * HQ_ + hq) * DK_;
    orow[lm]      = (__bf16)(o0[r] * inv);
    orow[16 + lm] = (__bf16)(o1[r] * inv);
    orow[32 + lm] = (__bf16)(o2[r] * inv);
    orow[48 + lm] = (__bf16)(o3[r] * inv);
  }
}

extern "C" void kernel_launch(void* const* d_in, const int* in_sizes, int n_in,
                              void* d_out, int out_size, void* d_ws, size_t ws_size,
                              hipStream_t stream) {
  (void)in_sizes; (void)n_in; (void)out_size; (void)ws_size;
  const float* x   = (const float*)d_in[0];
  const float* w_q = (const float*)d_in[1];
  const float* w_k = (const float*)d_in[2];
  const float* w_v = (const float*)d_in[3];
  const float* w_o = (const float*)d_in[4];
  float* out = (float*)d_out;

  char* ws = (char*)d_ws;
  size_t off = 0;
  auto alloc = [&](size_t bytes) { size_t p = off; off = (off + bytes + 255) & ~(size_t)255; return p; };

  __bf16* xh   = (__bf16*)(ws + alloc((size_t)BT_ * D_ * 2));
  __bf16* wqT  = (__bf16*)(ws + alloc((size_t)D_ * D_ * 2));
  __bf16* wkT  = (__bf16*)(ws + alloc((size_t)D_ * DKV_ * 2));
  __bf16* wvT  = (__bf16*)(ws + alloc((size_t)D_ * DKV_ * 2));
  __bf16* woT  = (__bf16*)(ws + alloc((size_t)D_ * D_ * 2));
  float*  qf   = (float*) (ws + alloc((size_t)BT_ * D_ * 4));
  float*  kf   = (float*) (ws + alloc((size_t)BT_ * DKV_ * 4));
  float*  vf   = (float*) (ws + alloc((size_t)BT_ * DKV_ * 4));
  __bf16* qbf  = (__bf16*)(ws + alloc((size_t)BT_ * D_ * 2));
  __bf16* kbf  = (__bf16*)(ws + alloc((size_t)BT_ * DKV_ * 2));
  __bf16* vtb  = (__bf16*)(ws + alloc((size_t)BT_ * DKV_ * 2));
  __bf16* attn = (__bf16*)(ws + alloc((size_t)BT_ * D_ * 2));

  const int TPB = 256;
  // 1) precision conversion / weight transposes
  f32_to_bf16<<<(BT_ * D_ + TPB - 1) / TPB, TPB, 0, stream>>>(x, xh, BT_ * D_);
  transpose_to_bf16<<<(D_ * D_ + TPB - 1) / TPB, TPB, 0, stream>>>(w_q, wqT, D_, D_);
  transpose_to_bf16<<<(D_ * DKV_ + TPB - 1) / TPB, TPB, 0, stream>>>(w_k, wkT, D_, DKV_);
  transpose_to_bf16<<<(D_ * DKV_ + TPB - 1) / TPB, TPB, 0, stream>>>(w_v, wvT, D_, DKV_);
  transpose_to_bf16<<<(D_ * D_ + TPB - 1) / TPB, TPB, 0, stream>>>(w_o, woT, D_, D_);

  // 2) Q/K/V projections (WMMA + async-LDS staging)
  dim3 gq(BT_ / 128, D_ / 64);
  dim3 gkv(BT_ / 128, DKV_ / 64);
  gemm_bf16_f32<<<gq,  TPB, 0, stream>>>(xh, wqT, qf, BT_, D_, D_);
  gemm_bf16_f32<<<gkv, TPB, 0, stream>>>(xh, wkT, kf, BT_, DKV_, D_);
  gemm_bf16_f32<<<gkv, TPB, 0, stream>>>(xh, wvT, vf, BT_, DKV_, D_);

  // 3) RoPE + layout conversions
  rope_convert<<<(B_ * HQ_ * T_ * 32 + TPB - 1) / TPB, TPB, 0, stream>>>(qf, qbf, HQ_);
  rope_convert<<<(B_ * HKV_ * T_ * 32 + TPB - 1) / TPB, TPB, 0, stream>>>(kf, kbf, HKV_);
  v_transpose<<<(B_ * HKV_ * DK_ * T_ + TPB - 1) / TPB, TPB, 0, stream>>>(vf, vtb);

  // 4) flash attention (WMMA, one wave per 16-row q tile)
  attn_fwd<<<(B_ * HQ_ * (T_ / 16)) / 8, TPB, 0, stream>>>(qbf, kbf, vtb, attn);

  // 5) output projection (WMMA) -> fp32 d_out
  gemm_bf16_f32<<<gq, TPB, 0, stream>>>(attn, woT, out, BT_, D_, D_);
}
